// GroupLoss_61486751809881
// MI455X (gfx1250) — matrix operand order repaired and checked by reading, not compiled, not run on hardware
//
#include <hip/hip_runtime.h>
#include <math.h>

// ---------------- CDNA5 WMMA types ----------------
typedef __attribute__((ext_vector_type(16))) __bf16 v16bf;
typedef __attribute__((ext_vector_type(8)))  float  v8f;
typedef int v4i __attribute__((vector_size(16)));  // matches async-LDS builtin param

union FragBF {
    v16bf v;
    unsigned u32[8];
    unsigned short u16[16];
};

__device__ inline v8f wmma_bf16(v16bf a, v16bf b, v8f c) {
    // D = A(16x32 bf16) * B(32x16 bf16) + C(16x16 f32)
    return __builtin_amdgcn_wmma_f32_16x16x32_bf16(
        /*neg_a=*/false, a, /*neg_b=*/false, b,
        /*c_mod=*/(short)0, c, /*reuse_a=*/false, /*reuse_b=*/false);
}

// float -> bf16 bits, round-to-nearest-even
__device__ inline unsigned short f2bf(float f) {
    unsigned u = __float_as_uint(f);
    u = (u + 0x7FFFu + ((u >> 16) & 1u)) >> 16;
    return (unsigned short)u;
}

// -------- CDNA5 async global->LDS copy (guarded; falls back cleanly) --------
#ifdef __has_builtin
#if __has_builtin(__builtin_amdgcn_global_load_async_to_lds_b128) && \
    __has_builtin(__builtin_amdgcn_s_wait_asynccnt)
#define USE_ASYNC_LDS 1
#endif
#endif
#ifndef USE_ASYNC_LDS
#define USE_ASYNC_LDS 0
#endif

#define AS_GLOBAL __attribute__((address_space(1)))
#define AS_LDS    __attribute__((address_space(3)))

// ---------------- block reductions (256 threads) ----------------
__device__ inline float block_reduce_sum256(float v, float* red) {
    int tid = threadIdx.x;
    red[tid] = v; __syncthreads();
    for (int s = 128; s > 0; s >>= 1) {
        if (tid < s) red[tid] += red[tid + s];
        __syncthreads();
    }
    float r = red[0]; __syncthreads();
    return r;
}
__device__ inline float block_reduce_max256(float v, float* red) {
    int tid = threadIdx.x;
    red[tid] = v; __syncthreads();
    for (int s = 128; s > 0; s >>= 1) {
        if (tid < s) red[tid] = fmaxf(red[tid], red[tid + s]);
        __syncthreads();
    }
    float r = red[0]; __syncthreads();
    return r;
}

// ---------------- prep kernels ----------------
__global__ __launch_bounds__(256) void convert_bf16_kernel(const float* __restrict__ in,
                                                           unsigned short* __restrict__ out,
                                                           size_t n) {
    size_t i = (size_t)blockIdx.x * blockDim.x + threadIdx.x;
    size_t stride = (size_t)gridDim.x * blockDim.x;
    for (; i < n; i += stride) out[i] = f2bf(in[i]);
}

// pad fc_w (c x d) into (cpad x d) bf16, zero-filled tail rows
__global__ __launch_bounds__(256) void pad_fcw_kernel(const float* __restrict__ fcw,
                                                      unsigned short* __restrict__ out,
                                                      int c, int cpad, int d) {
    size_t total = (size_t)cpad * d;
    size_t i = (size_t)blockIdx.x * blockDim.x + threadIdx.x;
    size_t stride = (size_t)gridDim.x * blockDim.x;
    for (; i < total; i += stride) {
        int row = (int)(i / d);
        int col = (int)(i % d);
        out[i] = (row < c) ? f2bf(fcw[(size_t)row * d + col]) : (unsigned short)0;
    }
}

// e = (emb - rowmean) / max(||.||, 1e-12)  -> bf16
__global__ __launch_bounds__(256) void center_norm_kernel(const float* __restrict__ emb,
                                                          unsigned short* __restrict__ e,
                                                          int d) {
    __shared__ float red[256];
    int row = blockIdx.x;
    const float* x = emb + (size_t)row * d;
    float s = 0.f, ss = 0.f;
    for (int i = threadIdx.x; i < d; i += 256) {
        float v = x[i];
        s += v; ss += v * v;
    }
    float tot  = block_reduce_sum256(s, red);
    float tots = block_reduce_sum256(ss, red);
    float mean = tot / (float)d;
    float var  = fmaxf(tots - (float)d * mean * mean, 0.f);
    float nrm  = fmaxf(sqrtf(var), 1e-12f);
    float inv  = 1.0f / nrm;
    unsigned short* er = e + (size_t)row * d;
    for (int i = threadIdx.x; i < d; i += 256)
        er[i] = f2bf((x[i] - mean) * inv);
}

// flags[perm[k]] = (k < n_select)
__global__ __launch_bounds__(256) void flags_kernel(const int* __restrict__ perm,
                                                    int* __restrict__ flags,
                                                    int n_select, int n) {
    int k = blockIdx.x * blockDim.x + threadIdx.x;
    if (k < n) flags[perm[k]] = (k < n_select) ? 1 : 0;
}

// Build X^T (cpad x n) bf16: onehot rows or softmax(logits+bias)
__global__ __launch_bounds__(256) void build_x_kernel(const float* __restrict__ logits,
                                                      const float* __restrict__ fcb,
                                                      const int* __restrict__ flags,
                                                      const int* __restrict__ lbs,
                                                      unsigned short* __restrict__ Xt,
                                                      int c, int cpad, int n) {
    __shared__ float red[256];
    int row = blockIdx.x;
    if (flags[row]) {
        int lab = lbs[row];
        for (int cc = threadIdx.x; cc < cpad; cc += 256)
            Xt[(size_t)cc * n + row] = (cc == lab) ? f2bf(1.0f) : (unsigned short)0;
        return;
    }
    const float* l = logits + (size_t)row * cpad;
    float mx = -3.402823466e+38f;
    for (int cc = threadIdx.x; cc < c; cc += 256)
        mx = fmaxf(mx, l[cc] + fcb[cc]);
    mx = block_reduce_max256(mx, red);
    float se = 0.f;
    for (int cc = threadIdx.x; cc < c; cc += 256)
        se += expf(l[cc] + fcb[cc] - mx);
    se = block_reduce_sum256(se, red);
    float inv = 1.0f / se;
    for (int cc = threadIdx.x; cc < cpad; cc += 256) {
        float p = (cc < c) ? expf(l[cc] + fcb[cc] - mx) * inv : 0.f;
        Xt[(size_t)cc * n + row] = f2bf(p);
    }
}

// Row-normalize Y (n x cpad) and write X^T bf16 for the next GEMM
__global__ __launch_bounds__(256) void normalize_kernel(const float* __restrict__ Y,
                                                        unsigned short* __restrict__ Xt,
                                                        int cpad, int n) {
    __shared__ float red[256];
    int row = blockIdx.x;
    const float* yr = Y + (size_t)row * cpad;
    float s = 0.f;
    for (int cc = threadIdx.x; cc < cpad; cc += 256) s += yr[cc];
    s = block_reduce_sum256(s, red) + 1e-6f;
    float inv = 1.0f / s;
    for (int cc = threadIdx.x; cc < cpad; cc += 256)
        Xt[(size_t)cc * n + row] = f2bf(yr[cc] * inv);
}

// Final iteration: per-row -log(X[row, lbs[row]]) after normalization
__global__ __launch_bounds__(256) void final_loss_kernel(const float* __restrict__ Y,
                                                         const int* __restrict__ lbs,
                                                         float* __restrict__ partial,
                                                         int cpad) {
    __shared__ float red[256];
    int row = blockIdx.x;
    const float* yr = Y + (size_t)row * cpad;
    float s = 0.f;
    for (int cc = threadIdx.x; cc < cpad; cc += 256) s += yr[cc];
    s = block_reduce_sum256(s, red) + 1e-6f;
    if (threadIdx.x == 0) {
        float v = yr[lbs[row]] / s;
        partial[row] = -logf(v);
    }
}

__global__ __launch_bounds__(256) void reduce_loss_kernel(const float* __restrict__ partial,
                                                          float* __restrict__ out, int n) {
    __shared__ float red[256];
    float s = 0.f;
    for (int i = threadIdx.x; i < n; i += 256) s += partial[i];
    s = block_reduce_sum256(s, red);
    if (threadIdx.x == 0) out[0] = s / (float)n;
}

// ---------------- WMMA GEMM: C(MxN) = A(MxK) * B(NxK)^T ----------------
// A,B bf16 row-major (K contiguous). 256 threads = 8 wave32s (4x2); block
// tile 128x128, each wave computes 32x64 = 2x4 WMMA 16x16 tiles; K-step 32.
// LDS tiles are double-buffered -> single barrier per K-step, and staged with
// GLOBAL_LOAD_ASYNC_TO_LDS_B128 (ASYNCcnt) when the builtin is available.
// MODE 0: store f32 C. MODE 1: relu + zero global diagonal, store bf16 C.
#define BMT 128
#define BNT 128
#define BKT 32
#define LDT 48  // padded LDS row stride in bf16 elems (96B, 16B-aligned)

template <int MODE>
__global__ __launch_bounds__(256) void gemm_nt_wmma(const unsigned short* __restrict__ A, int lda,
                                                    const unsigned short* __restrict__ B, int ldb,
                                                    float* __restrict__ Cf,
                                                    unsigned short* __restrict__ Cb,
                                                    int ldc, int K) {
    __shared__ __align__(16) unsigned short As[2][BMT][LDT];
    __shared__ __align__(16) unsigned short Bs[2][BNT][LDT];

    const int bm   = blockIdx.y * BMT;
    const int bn   = blockIdx.x * BNT;
    const int tid  = threadIdx.x;
    const int wave = tid >> 5;     // 0..7
    const int lane = tid & 31;
    const int wm   = wave >> 1;    // 0..3 : M sub-tile (x32)
    const int wn   = wave & 1;     // 0..1 : N sub-tile (x64)
    const int half = lane >> 4;    // 0/1
    const int r    = lane & 15;

    // stage one 128x32 bf16 tile of A and of B into LDS buffer `buf`
    auto stage = [&](int buf, int kk) {
#pragma unroll
        for (int i = 0; i < 2; ++i) {
            int idx  = tid + i * 256;   // 0..511
            int row  = idx >> 2;
            int colv = idx & 3;         // which uint4 (8 bf16) within the row
            const unsigned short* sa = A + (size_t)(bm + row) * lda + kk + colv * 8;
            const unsigned short* sb = B + (size_t)(bn + row) * ldb + kk + colv * 8;
#if USE_ASYNC_LDS
            __builtin_amdgcn_global_load_async_to_lds_b128(
                (AS_GLOBAL v4i*)sa, (AS_LDS v4i*)&As[buf][row][colv * 8], 0, 0);
            __builtin_amdgcn_global_load_async_to_lds_b128(
                (AS_GLOBAL v4i*)sb, (AS_LDS v4i*)&Bs[buf][row][colv * 8], 0, 0);
#else
            *(uint4*)&As[buf][row][colv * 8] = *(const uint4*)sa;
            *(uint4*)&Bs[buf][row][colv * 8] = *(const uint4*)sb;
#endif
        }
    };

    v8f zero8 = {0.f, 0.f, 0.f, 0.f, 0.f, 0.f, 0.f, 0.f};
    v8f acc[2][4] = {{zero8, zero8, zero8, zero8}, {zero8, zero8, zero8, zero8}};

    stage(0, 0);
#if USE_ASYNC_LDS
    __builtin_amdgcn_s_wait_asynccnt(0);
#endif
    __syncthreads();

    int p = 0;
    for (int k0 = 0; k0 < K; k0 += BKT) {
        // stage next K slab into the other buffer (overlaps with WMMA below)
        if (k0 + BKT < K) stage(p ^ 1, k0 + BKT);
#if !USE_ASYNC_LDS
        if (k0 + 2 * BKT < K) {  // uniform-branch cache prefetch two slabs ahead
            __builtin_prefetch(A + (size_t)(bm + (tid >> 1)) * lda + k0 + 2 * BKT, 0, 0);
            __builtin_prefetch(B + (size_t)(bn + (tid >> 1)) * ldb + k0 + 2 * BKT, 0, 0);
        }
#endif
        // ---- gather fragments from buffer p per CDNA5 wave32 WMMA layouts ----
        FragBF a[2], b[4];
        const unsigned* arow0 = (const unsigned*)As[p][wm * 32 + r];
        const unsigned* arow1 = (const unsigned*)As[p][wm * 32 + 16 + r];
#pragma unroll
        for (int v = 0; v < 8; ++v) {
            // A 16x32: VGPR v holds K pair ((v&4)?16:0) + half*8 + (v&3)*2
            int ka2 = ((v & 4) ? 8 : 0) + half * 4 + (v & 3);  // u32 index
            a[0].u32[v] = arow0[ka2];
            a[1].u32[v] = arow1[ka2];
        }
#pragma unroll
        for (int t = 0; t < 4; ++t) {
            // B 32x16: VGPR v holds K pair ((v&4)?16:0) + half*8 + (v&3)*2 (matches A)
            const unsigned* brow = (const unsigned*)Bs[p][wn * 64 + t * 16 + r];
#pragma unroll
            for (int v = 0; v < 8; ++v) b[t].u32[v] = brow[((v & 4) ? 8 : 0) + half * 4 + (v & 3)];
        }

#pragma unroll
        for (int mi = 0; mi < 2; ++mi)
#pragma unroll
            for (int ni = 0; ni < 4; ++ni)
                acc[mi][ni] = wmma_bf16(a[mi].v, b[ni].v, acc[mi][ni]);

#if USE_ASYNC_LDS
        __builtin_amdgcn_s_wait_asynccnt(0);  // own async copies landed in LDS
#endif
        __syncthreads();                      // everyone's copies landed / reads done
        p ^= 1;
    }

    // ---- epilogue: C layout -> lane(r,half), VGPR i = row half*8+i, col r ----
#pragma unroll
    for (int mi = 0; mi < 2; ++mi) {
#pragma unroll
        for (int ni = 0; ni < 4; ++ni) {
            int gr0 = bm + wm * 32 + mi * 16 + half * 8;
            int gc  = bn + wn * 64 + ni * 16 + r;
#pragma unroll
            for (int i = 0; i < 8; ++i) {
                float val = acc[mi][ni][i];
                int gr = gr0 + i;
                if (MODE == 1) {
                    val = fmaxf(val, 0.f);
                    if (gr == gc) val = 0.f;
                    Cb[(size_t)gr * ldc + gc] = f2bf(val);
                } else {
                    Cf[(size_t)gr * ldc + gc] = val;
                }
            }
        }
    }
}

// ---------------- host launcher ----------------
extern "C" void kernel_launch(void* const* d_in, const int* in_sizes, int n_in,
                              void* d_out, int out_size, void* d_ws, size_t ws_size,
                              hipStream_t stream) {
    const int n = 8192, d = 2048, c = 1000, cp = 1024;
    const int n_select = c * 2;  // N_LBS_PER_CLS = 2

    const float* emb  = (const float*)d_in[0];
    const float* fc_w = (const float*)d_in[1];
    const float* fc_b = (const float*)d_in[2];
    const int*   lbs  = (const int*)d_in[3];
    const int*   perm = (const int*)d_in[4];
    float* out = (float*)d_out;

    // workspace carve-up (256B aligned)
    char* p = (char*)d_ws;
    auto carve = [&](size_t bytes) -> void* {
        void* r = (void*)p;
        p += (bytes + 255) & ~(size_t)255;
        return r;
    };
    unsigned short* emb_bf = (unsigned short*)carve((size_t)n * d * 2);
    unsigned short* fcw_bf = (unsigned short*)carve((size_t)cp * d * 2);
    unsigned short* e_bf   = (unsigned short*)carve((size_t)n * d * 2);
    float*          logits = (float*)carve((size_t)n * cp * 4);
    int*            flags  = (int*)carve((size_t)n * 4);
    unsigned short* Xt0    = (unsigned short*)carve((size_t)cp * n * 2);
    unsigned short* Xt1    = (unsigned short*)carve((size_t)cp * n * 2);
    unsigned short* Wb     = (unsigned short*)carve((size_t)n * n * 2);
    float*          Y      = (float*)carve((size_t)n * cp * 4);
    float*          part   = (float*)carve((size_t)n * 4);

    // 1. bf16 conversions
    convert_bf16_kernel<<<4096, 256, 0, stream>>>(emb, emb_bf, (size_t)n * d);
    pad_fcw_kernel<<<4096, 256, 0, stream>>>(fc_w, fcw_bf, c, cp, d);

    // 2. e = normalize(center(emb))
    center_norm_kernel<<<n, 256, 0, stream>>>(emb, e_bf, d);

    // 3. logits = emb @ fc_w^T  (M=8192, N=1024, K=2048)
    gemm_nt_wmma<0><<<dim3(cp / BNT, n / BMT), 256, 0, stream>>>(
        emb_bf, d, fcw_bf, d, logits, nullptr, cp, d);

    // 4. onehot/prob flags, then build X^T (softmax + bias / one-hot)
    flags_kernel<<<(n + 255) / 256, 256, 0, stream>>>(perm, flags, n_select, n);
    build_x_kernel<<<n, 256, 0, stream>>>(logits, fc_b, flags, lbs, Xt0, c, cp, n);

    // 5. W = relu(zero-diag(e @ e^T)) bf16  (M=N=8192, K=2048) — L2-resident
    gemm_nt_wmma<1><<<dim3(n / BNT, n / BMT), 256, 0, stream>>>(
        e_bf, d, e_bf, d, nullptr, Wb, n, d);

    // 6. iteration 1: Y = W @ X0 ; row-normalize -> Xt1
    gemm_nt_wmma<0><<<dim3(cp / BNT, n / BMT), 256, 0, stream>>>(
        Wb, n, Xt0, n, Y, nullptr, cp, n);
    normalize_kernel<<<n, 256, 0, stream>>>(Y, Xt1, cp, n);

    // 7. iteration 2: Y = W @ X1 ; normalize + gather per-row loss
    gemm_nt_wmma<0><<<dim3(cp / BNT, n / BMT), 256, 0, stream>>>(
        Wb, n, Xt1, n, Y, nullptr, cp, n);
    final_loss_kernel<<<n, 256, 0, stream>>>(Y, lbs, part, cp);

    // 8. deterministic tree reduction -> scalar loss
    reduce_loss_kernel<<<1, 256, 0, stream>>>(part, out, n);
}
